// Yolov8ObjectDetector_28260884808417
// MI455X (gfx1250) — compile-verified, matrix operand order checked by reading, not run on hardware
//
#include <hip/hip_runtime.h>
#include <stdint.h>

// ---- problem constants (match reference) ----
#define NCLS   80
#define NROWS  84          // 4 box rows + 80 class rows
#define NA     8400        // anchors
#define NBATCH 32
#define NPAD   16384       // anchors padded to power of two for selection
#define KCAND  1024        // pre-NMS candidate cap
#define MAXDET 300
#define CONF_T 0.25f
#define IOU_THR 0.45f

typedef __attribute__((ext_vector_type(2)))  float    v2f;
typedef __attribute__((ext_vector_type(8)))  float    v8f;
typedef __attribute__((ext_vector_type(16))) _Float16 v16h;
typedef __attribute__((ext_vector_type(4)))  unsigned v4u;
typedef __attribute__((ext_vector_type(4)))  int      v4i;
typedef __attribute__((ext_vector_type(8)))  int      v8i;

#if defined(__has_builtin)
#  if __has_builtin(__builtin_amdgcn_tensor_load_to_lds) && \
      __has_builtin(__builtin_amdgcn_s_wait_tensorcnt)
#    define USE_TDM 1
#  endif
#  if __has_builtin(__builtin_amdgcn_ballot_w32)
#    define USE_BALLOT 1
#  endif
#endif
#ifndef USE_TDM
#  define USE_TDM 0
#endif
#ifndef USE_BALLOT
#  define USE_BALLOT 0
#endif

// =====================================================================
// Kernel 1: per-anchor best-class confidence + argmax, build sort keys.
// key = (score_bits << 32) | (0xFFFFFFFF - anchor)  -> descending u64
// gives descending score with ties broken by LOWER anchor index,
// matching jax.lax.top_k (score >= 0 so float bits order correctly).
// =====================================================================
__global__ __launch_bounds__(256)
void yolo_score_kernel(const float* __restrict__ preds,
                       unsigned long long* __restrict__ keys,
                       int* __restrict__ clsbuf) {
  const int n = blockIdx.x * 256 + threadIdx.x;   // padded anchor id
  const int b = blockIdx.y;
  unsigned sbits = 0u;
  if (n < NA) {
    const float* p = preds + (size_t)b * NROWS * NA + 4 * NA + n;
    __builtin_prefetch(p, 0, 1);                  // global_prefetch_b8
    __builtin_prefetch(p + 40 * NA, 0, 1);
    float best = -1.0f; int bc = 0;
#pragma unroll 4
    for (int c = 0; c < NCLS; ++c) {              // coalesced across lanes
      float v = p[(size_t)c * NA];
      if (v > best) { best = v; bc = c; }         // first-max like argmax
    }
    float score = (best > CONF_T) ? best : 0.0f;
    sbits = __float_as_uint(score);
    clsbuf[b * NA + n] = bc;
  }
  if (n < NPAD) {
    keys[(size_t)b * NPAD + n] =
        ((unsigned long long)sbits << 32) |
        (unsigned long long)(0xFFFFFFFFu - (unsigned)n);
  }
}

// low 32 bits of a generic pointer to LDS == byte offset in LDS aperture
__device__ __forceinline__ unsigned lds_off(const void* p) {
  return (unsigned)(uintptr_t)p;
}

#if USE_TDM
// TDM: 1D tile of `nelem` u64 elements, global -> LDS (D# per ISA 08 §8).
__device__ __forceinline__ void tdm_load_1d_u64(unsigned lds_byte_off,
                                                const void* gptr,
                                                unsigned nelem) {
  unsigned long long ga = (unsigned long long)(uintptr_t)gptr;
  v4u g0;
  g0[0] = 1u;                                        // count=1, user descriptor
  g0[1] = lds_byte_off;                              // lds_addr
  g0[2] = (unsigned)(ga & 0xFFFFFFFFu);              // global_addr[31:0]
  g0[3] = (unsigned)((ga >> 32) & 0x01FFFFFFu)       // global_addr[56:32]
          | (2u << 30);                              // type=2 ("image")
  v8i g1;
  g1[0] = (int)(3u << 16);                           // data_size=3 (8B), mask=0
  g1[1] = (int)((nelem & 0xFFFFu) << 16);            // tensor_dim0[15:0]
  g1[2] = (int)(((nelem >> 16) & 0xFFFFu)            // tensor_dim0[31:16]
          | (1u << 16));                             // tensor_dim1 = 1
  g1[3] = (int)((nelem & 0xFFFFu) << 16);            // tile_dim0
  g1[4] = 0;                                         // tile_dim1/2 unused
  g1[5] = (int)nelem;                                // tensor_dim0_stride lo
  g1[6] = 0;
  g1[7] = 0;
  v4i z4 = {0, 0, 0, 0};
#if __clang_major__ >= 23
  v8i z8 = {0, 0, 0, 0, 0, 0, 0, 0};
  __builtin_amdgcn_tensor_load_to_lds(g0, g1, z4, z4, z8, 0);
#else
  __builtin_amdgcn_tensor_load_to_lds(g0, g1, z4, z4, 0);
#endif
}
#endif

// descending bitonic sort of 1024 u64 keys in LDS, blockDim.x == 1024
__device__ __forceinline__ void bitonic_sort_desc_1024(unsigned long long* arr,
                                                       unsigned tid) {
  for (unsigned k = 2; k <= KCAND; k <<= 1) {
    for (unsigned j = k >> 1; j > 0; j >>= 1) {
      unsigned i = tid, ixj = i ^ j;
      if (ixj > i) {
        unsigned long long a = arr[i], c = arr[ixj];
        bool up = ((i & k) == 0);                 // descending segments
        bool sw = up ? (a < c) : (a > c);
        if (sw) { arr[i] = c; arr[ixj] = a; }
      }
      __syncthreads();
    }
  }
}

// =====================================================================
// Kernel 2: one workgroup (1024 thr = 32 waves) per batch item.
//  phase 1: streaming bitonic top-1024 over 16 chunks; chunks staged
//           into LDS by the Tensor Data Mover, double-buffered, with
//           s_wait_tensorcnt overlap (DMA of chunk c+1 under sort of c)
//  phase 2: gather boxes; 1024x1024 suppression bitmask; the
//           (area_i + area_j) matrix from V_WMMA_F32_16X16X4_F32;
//           iou > T tested divide-free as inter > T*(S-inter+eps);
//           bits aggregated per row with wave32 ballot -> 1 atomic
//  phase 3: serial greedy NMS reduced to 1024 x (32-word OR)
//  phase 4: emit first <=300 kept rows, pad with NCLS
// =====================================================================
__global__ __launch_bounds__(1024)
void yolo_nms_kernel(const float* __restrict__ preds,
                     const unsigned long long* __restrict__ keys,
                     const int* __restrict__ clsbuf,
                     unsigned int* __restrict__ supmat,   // NBATCH*KCAND*32
                     float* __restrict__ out) {
  const unsigned tid  = threadIdx.x;
  const int      b    = blockIdx.x;
  const unsigned lane = tid & 31u;
  const unsigned wave = tid >> 5;

  __shared__ unsigned long long s_res[KCAND];
  __shared__ unsigned long long s_bufA[KCAND];
  __shared__ unsigned long long s_bufB[KCAND];
  __shared__ float s_x1[KCAND], s_y1[KCAND], s_x2[KCAND], s_y2[KCAND];
  __shared__ float s_area[KCAND], s_score[KCAND];
  __shared__ int   s_cls[KCAND];
  __shared__ unsigned s_supp[KCAND / 32];
  __shared__ int   s_kept[MAXDET];
  __shared__ int   s_nkept;

  const unsigned long long* gk = keys + (size_t)b * NPAD;

  // ---------------- phase 1: top-1024 (exact, sorted desc) ------------
  // s_res starts all-zero: every real key > 0, zeros fully displaced.
  s_res[tid] = 0ull;
#if USE_TDM
  if (tid < 32) tdm_load_1d_u64(lds_off(s_bufA), gk, KCAND);   // chunk 0
#endif
  __syncthreads();
  for (int c = 0; c < NPAD / KCAND; ++c) {
    unsigned long long* cur = (c & 1) ? s_bufB : s_bufA;
#if USE_TDM
    unsigned long long* nxt = (c & 1) ? s_bufA : s_bufB;
    if (tid < 32) {
      if (c + 1 < NPAD / KCAND) {
        tdm_load_1d_u64(lds_off(nxt), gk + (size_t)(c + 1) * KCAND, KCAND);
        __builtin_amdgcn_s_wait_tensorcnt(1);   // chunk c retired (in-order)
      } else {
        __builtin_amdgcn_s_wait_tensorcnt(0);
      }
    }
    __syncthreads();
#else
    cur[tid] = gk[(size_t)c * KCAND + tid];
    __syncthreads();
#endif
    bitonic_sort_desc_1024(cur, tid);
    // bitonic merge step j=1024: keep max half (res desc ++ rev(cur) asc)
    unsigned long long a = s_res[tid];
    unsigned long long d = cur[(KCAND - 1) - tid];
    s_res[tid] = (a > d) ? a : d;
    __syncthreads();
    // cleanup: descending bitonic merge of the surviving 1024
    for (unsigned j = KCAND >> 1; j > 0; j >>= 1) {
      unsigned i = tid, ixj = i ^ j;
      if (ixj > i) {
        unsigned long long x = s_res[i], y = s_res[ixj];
        if (x < y) { s_res[i] = y; s_res[ixj] = x; }
      }
      __syncthreads();
    }
  }

  // ---------------- phase 2a: gather + box convert --------------------
  {
    unsigned long long key = s_res[tid];
    float score  = __uint_as_float((unsigned)(key >> 32));
    unsigned idx = 0xFFFFFFFFu - (unsigned)(key & 0xFFFFFFFFull);
    unsigned src = (idx < NA) ? idx : 0u;        // padded entries: score==0
    const float* pb = preds + (size_t)b * NROWS * NA;
    float x = pb[0 * NA + src], y = pb[1 * NA + src];
    float w = pb[2 * NA + src], h = pb[3 * NA + src];
    float hw = w * 0.5f, hh = h * 0.5f;
    float x1 = x - hw, y1 = y - hh, x2 = x + hw, y2 = y + hh;
    s_x1[tid] = x1; s_y1[tid] = y1; s_x2[tid] = x2; s_y2[tid] = y2;
    s_area[tid]  = (x2 - x1) * (y2 - y1);        // same fp path as reference
    s_score[tid] = score;
    s_cls[tid]   = clsbuf[b * NA + src];
  }
  unsigned* gsup = supmat + (size_t)b * KCAND * 32;
  for (unsigned wd = tid; wd < KCAND * 32; wd += 1024) gsup[wd] = 0u;
  __threadfence();
  __syncthreads();

  // ---------------- phase 2b: suppression bitmask via WMMA ------------
  // 64x64 tiles of 16x16; 32 waves -> 128 tiles per wave.
  for (int t = (int)wave; t < 64 * 64; t += 32) {
    int ti = t >> 6, tj = t & 63;
    bool lo = lane < 16;
    int  lr = (int)(lane & 15);
    float aA   = lo ? s_area[ti * 16 + lr] : 0.0f;   // A rows [area_i, 1, 0, 0]
    float aB   = lo ? s_area[tj * 16 + lr] : 0.0f;   // B cols [1, area_j, 0, 0]^T
    float oneV = lo ? 1.0f : 0.0f;
    v8f c = {};
#if __has_builtin(__builtin_amdgcn_wmma_f32_16x16x4_f32)
    v2f af, bf;
    af.x = aA;   af.y = oneV;            // lanes 0-15: K0,K1; hi lanes: K2,K3=0
    bf.x = oneV; bf.y = aB;
    c = __builtin_amdgcn_wmma_f32_16x16x4_f32(false, af, false, bf,
                                              (short)0, c, false, false);
#else
    v16h ah = {}, bh = {};
    ah[0] = (_Float16)aA;   ah[1] = (_Float16)oneV;
    bh[0] = (_Float16)oneV; bh[1] = (_Float16)aB;
    c = __builtin_amdgcn_wmma_f32_16x16x32_f16(false, ah, false, bh,
                                               (short)0, c, false, false);
#endif
    // c[v] = area_i + area_j at (M = v + 8*(lane>=16), N = lane&15)
    int ibase = ti * 16 + (int)((lane >> 4) << 3);
    int j     = tj * 16 + lr;
    float jx1 = s_x1[j], jy1 = s_y1[j], jx2 = s_x2[j], jy2 = s_y2[j];
    int   jc  = s_cls[j];
#pragma unroll
    for (int v = 0; v < 8; ++v) {
      int i = ibase + v;
      float lx = fmaxf(s_x1[i], jx1), ly = fmaxf(s_y1[i], jy1);
      float rx = fminf(s_x2[i], jx2), ry = fminf(s_y2[i], jy2);
      float iw = fmaxf(rx - lx, 0.0f), ih = fmaxf(ry - ly, 0.0f);
      float inter = iw * ih;
      // union > 0 always => iou > T  <=>  inter > T * union  (divide-free)
      bool bit = (j > i) && (s_cls[i] == jc) &&
                 (inter > IOU_THR * (c[v] - inter + 1e-7f));
#if USE_BALLOT
      unsigned m = __builtin_amdgcn_ballot_w32(bit);
      // bits 0-15: row ti*16+v ; bits 16-31: row ti*16+8+v ; cols tj*16..+15
      unsigned loh = m & 0xFFFFu, hih = m >> 16;
      unsigned sh  = (unsigned)(tj & 1) * 16u;
      if (lane == 0 && loh)
        atomicOr(&gsup[(ti * 16 + v) * 32 + (tj >> 1)], loh << sh);
      if (lane == 16 && hih)
        atomicOr(&gsup[(ti * 16 + 8 + v) * 32 + (tj >> 1)], hih << sh);
#else
      if (bit) atomicOr(&gsup[i * 32 + (j >> 5)], 1u << (j & 31));
#endif
    }
  }
  __threadfence();
  __syncthreads();

  // ---------------- phase 3: greedy NMS (serial, tiny per step) -------
  if (tid == 0) s_nkept = 0;
  if (tid < 32) s_supp[tid] = 0u;
  __syncthreads();
  for (int i = 0; i < KCAND; ++i) {
    bool ki = (s_score[i] > CONF_T) &&
              !((s_supp[i >> 5] >> (i & 31)) & 1u);
    __syncthreads();
    if (ki) {
      if (tid < 32) s_supp[tid] |= gsup[i * 32 + tid];
      if (tid == 0) {
        int n = s_nkept;
        if (n < MAXDET) s_kept[n] = i;
        s_nkept = n + 1;
      }
    }
    __syncthreads();
  }

  // ---------------- phase 4: emit [B, 300, 6] -------------------------
  if (tid < MAXDET) {
    int nk = (s_nkept < MAXDET) ? s_nkept : MAXDET;
    float r0, r1, r2, r3, r4, r5;
    if ((int)tid < nk) {
      int i = s_kept[tid];
      r0 = s_x1[i]; r1 = s_y1[i]; r2 = s_x2[i]; r3 = s_y2[i];
      r4 = s_score[i]; r5 = (float)s_cls[i];
    } else {
      r0 = r1 = r2 = r3 = r4 = r5 = (float)NCLS;   // pad value = 80
    }
    float* o = out + ((size_t)b * MAXDET + tid) * 6;
    o[0] = r0; o[1] = r1; o[2] = r2; o[3] = r3; o[4] = r4; o[5] = r5;
  }
}

// =====================================================================
extern "C" void kernel_launch(void* const* d_in, const int* in_sizes, int n_in,
                              void* d_out, int out_size, void* d_ws, size_t ws_size,
                              hipStream_t stream) {
  (void)in_sizes; (void)n_in; (void)out_size; (void)ws_size;
  const float* preds = (const float*)d_in[0];     // [32, 84, 8400] f32
  float* out = (float*)d_out;                     // [32, 300, 6]   f32

  // workspace layout (~9.2 MB total)
  unsigned long long* keys = (unsigned long long*)d_ws;          // 32*16384 u64
  int* clsbuf   = (int*)(keys + (size_t)NBATCH * NPAD);          // 32*8400  i32
  unsigned* sup = (unsigned*)(clsbuf + (size_t)NBATCH * NA);     // 32*32768 u32

  dim3 gScore(NPAD / 256, NBATCH);
  yolo_score_kernel<<<gScore, dim3(256), 0, stream>>>(preds, keys, clsbuf);
  yolo_nms_kernel<<<dim3(NBATCH), dim3(1024), 0, stream>>>(preds, keys, clsbuf,
                                                           sup, out);
}
